// GAT_33054068310176
// MI455X (gfx1250) — matrix operand order repaired and checked by reading, not compiled, not run on hardware
//
#include <hip/hip_runtime.h>
#include <hip/hip_bf16.h>

typedef __attribute__((ext_vector_type(16))) _Float16 v16h;
typedef __attribute__((ext_vector_type(8)))  float    v8f;
typedef __attribute__((__vector_size__(16))) int      v4i;

#define GN   8192
#define GFT  512
#define GC   64
#define ALPHA 0.2f

#if __has_builtin(__builtin_amdgcn_global_load_async_to_lds_b128) && \
    __has_builtin(__builtin_amdgcn_s_wait_asynccnt)
#define USE_ASYNC_LDS 1
#else
#define USE_ASYNC_LDS 0
#endif

#define AS1 __attribute__((address_space(1)))
#define AS3 __attribute__((address_space(3)))

__device__ __forceinline__ float leaky(float x) {
    return fmaxf(x, 0.f) + ALPHA * fminf(x, 0.f);
}

// ---------------------------------------------------------------------------
// Kernel 0: Wt[c][k] = (f16) W[k][c]   (64 x 512, 64KB, L2-resident)
// ---------------------------------------------------------------------------
__global__ void wt_kernel(const float* __restrict__ W, _Float16* __restrict__ Wt) {
    int c = blockIdx.x;
    for (int k = threadIdx.x; k < GFT; k += 256)
        Wt[(size_t)c * GFT + k] = (_Float16)W[(size_t)k * GC + c];
}

// ---------------------------------------------------------------------------
// Kernel 1: Wh = features @ W via f16 WMMA, f32 accumulate.
// LDS tile in B-operand order [c][k]: each B fragment = contiguous 32B read.
// Writes Wh (f32) and WhT (f16 column-major) for the attention GEMM.
// ---------------------------------------------------------------------------
__global__ __launch_bounds__(32)
void wh_kernel(const float* __restrict__ F, const _Float16* __restrict__ Wt,
               float* __restrict__ Wh, _Float16* __restrict__ WhT) {
    __shared__ __align__(64) _Float16 ldsK[GC * 32];   // [c][k] 4KB
    const int lane    = threadIdx.x;                   // 0..31 (wave32)
    const int rowBase = blockIdx.x * 16;
    const int m       = lane & 15;
    const int kbA     = (lane < 16) ? 0 : 16;          // A 16x32 f16 layout
    const int kbB     = (lane < 16) ? 0 : 16;          // B 32x16 f16 layout

    v8f zero = {0.f,0.f,0.f,0.f,0.f,0.f,0.f,0.f};
    v8f acc[4];
    #pragma unroll
    for (int t = 0; t < 4; ++t) acc[t] = zero;

    for (int k0 = 0; k0 < GFT; k0 += 32) {
        #pragma unroll
        for (int cc = 0; cc < 2; ++cc) {
            int c = lane * 2 + cc;
            const uint4* p = (const uint4*)(Wt + (size_t)c * GFT + k0);
            uint4* d = (uint4*)&ldsK[c * 32];
            d[0] = p[0]; d[1] = p[1]; d[2] = p[2]; d[3] = p[3];
        }
        __syncthreads();

        const float* frow = F + (size_t)(rowBase + m) * GFT + k0 + kbA;
        float4 fa = *(const float4*)(frow +  0);
        float4 fb = *(const float4*)(frow +  4);
        float4 fc = *(const float4*)(frow +  8);
        float4 fd = *(const float4*)(frow + 12);
        v16h A;
        A[0]=(_Float16)fa.x; A[1]=(_Float16)fa.y; A[2]=(_Float16)fa.z; A[3]=(_Float16)fa.w;
        A[4]=(_Float16)fb.x; A[5]=(_Float16)fb.y; A[6]=(_Float16)fb.z; A[7]=(_Float16)fb.w;
        A[8]=(_Float16)fc.x; A[9]=(_Float16)fc.y; A[10]=(_Float16)fc.z; A[11]=(_Float16)fc.w;
        A[12]=(_Float16)fd.x; A[13]=(_Float16)fd.y; A[14]=(_Float16)fd.z; A[15]=(_Float16)fd.w;

        #pragma unroll
        for (int t = 0; t < 4; ++t) {
            v16h B = *(const v16h*)&ldsK[(t * 16 + m) * 32 + kbB];
            acc[t] = __builtin_amdgcn_wmma_f32_16x16x32_f16(
                false, A, false, B, (short)0, acc[t], false, false);
        }
        __syncthreads();
    }

    const int mhi = (lane < 16) ? 0 : 8;
    #pragma unroll
    for (int t = 0; t < 4; ++t) {
        #pragma unroll
        for (int r = 0; r < 8; ++r) {
            int row = rowBase + r + mhi;
            int col = t * 16 + m;
            float v = acc[t][r];
            Wh[(size_t)row * GC + col] = v;
            WhT[(size_t)col * GN + row] = (_Float16)v;
        }
    }
}

// ---------------------------------------------------------------------------
// Kernel 2: src/dst rank-1 terms.
// ---------------------------------------------------------------------------
__global__ void srcdst_kernel(const float* __restrict__ Wh, const float* __restrict__ a,
                              float* __restrict__ src, float* __restrict__ dst) {
    int i = blockIdx.x * blockDim.x + threadIdx.x;
    const float* row = Wh + (size_t)i * GC;
    float s = 0.f, d = 0.f;
    #pragma unroll
    for (int c = 0; c < GC; ++c) {
        float v = row[c];
        s += v * a[c];
        d += v * a[GC + c];
    }
    src[i] = s;
    dst[i] = d;
}

// ---------------------------------------------------------------------------
// Kernel 3: global max of dst (softmax shift upper bound; shift-invariant).
// ---------------------------------------------------------------------------
__global__ void gmax_kernel(const float* __restrict__ dst, float* __restrict__ gmax) {
    __shared__ float red[256];
    float mv = -3.402823e38f;
    for (int i = threadIdx.x; i < GN; i += 256) mv = fmaxf(mv, dst[i]);
    red[threadIdx.x] = mv;
    __syncthreads();
    for (int s = 128; s > 0; s >>= 1) {
        if (threadIdx.x < s) red[threadIdx.x] = fmaxf(red[threadIdx.x], red[threadIdx.x + s]);
        __syncthreads();
    }
    if (threadIdx.x == 0) *gmax = red[0];
}

// ---------------------------------------------------------------------------
// Kernel 4: fused masked-softmax attention + (attn @ Wh) + elu.
// Single streaming pass over adj; p-tiles feed v_wmma_f32_16x16x32_f16.
// Wh tiles double-buffered in LDS via async global->LDS (ASYNCcnt) when
// available; register double-buffer otherwise. One barrier per j-tile.
// ---------------------------------------------------------------------------
__global__ __launch_bounds__(128)
void attn_kernel(const int* __restrict__ adjA, const int* __restrict__ adjB,
                 const _Float16* __restrict__ WhT,
                 const float* __restrict__ src, const float* __restrict__ dst,
                 const float* __restrict__ gmaxp, float* __restrict__ hout) {
    __shared__ __align__(64) _Float16 ldsT[2][GC * 32];   // [buf][c][k] 2 x 4KB
    __shared__ float ldsSum[4][16];

    const int tid   = threadIdx.x;
    const int wave  = tid >> 5;
    const int lane  = tid & 31;
    const int layer = blockIdx.y;
    const int* adj  = layer ? adjB : adjA;

    const int rowBase = (blockIdx.x * 4 + wave) * 16;
    const int m   = lane & 15;
    const int kbA = (lane < 16) ? 0 : 16;
    const int kbB = (lane < 16) ? 0 : 16;
    const int row = rowBase + m;

    const float srcv = src[row];
    const float gmax = *gmaxp;
    const float mval = leaky(srcv + gmax);      // >= leaky(src+dst) for all j

    v8f zero = {0.f,0.f,0.f,0.f,0.f,0.f,0.f,0.f};
    v8f acc[4];
    #pragma unroll
    for (int t = 0; t < 4; ++t) acc[t] = zero;
    float sumw = 0.f;

    const int* adjrow = adj + (size_t)row * GN;

    // Tile-copy role: thread tid handles column c_cp, k-halves [ko_cp, ko_cp+16)
    const int c_cp  = tid >> 1;
    const int ko_cp = (tid & 1) * 16;
    const _Float16* gsrc = WhT + (size_t)c_cp * GN + ko_cp;

#if USE_ASYNC_LDS
    // Preload tile 0 directly into LDS (no VGPR round-trip).
    {
        AS1 v4i* gp = (AS1 v4i*)(AS1 char*)(_Float16*)gsrc;
        AS3 v4i* lp = (AS3 v4i*)(AS3 char*)&ldsT[0][c_cp * 32 + ko_cp];
        __builtin_amdgcn_global_load_async_to_lds_b128(gp, lp, 0, 0);
        __builtin_amdgcn_global_load_async_to_lds_b128(gp, lp, 16, 0);
    }
    __builtin_amdgcn_s_wait_asynccnt(0);
    __syncthreads();
#else
    {
        uint4 r0 = *(const uint4*)(gsrc + 0);
        uint4 r1 = *(const uint4*)(gsrc + 8);
        uint4* d = (uint4*)&ldsT[0][c_cp * 32 + ko_cp];
        d[0] = r0; d[1] = r1;
    }
#endif

    int buf = 0;
    for (int jt = 0; jt < GN / 32; ++jt) {
        const int j = jt * 32;
        const bool more = (jt + 1) < (GN / 32);

#if USE_ASYNC_LDS
        // Issue next tile into buf^1; its previous readers finished before the
        // barrier that ended the last iteration. Copy overlaps compute below.
        if (more) {
            AS1 v4i* gp = (AS1 v4i*)(AS1 char*)(_Float16*)(gsrc + (size_t)(j + 32));
            AS3 v4i* lp = (AS3 v4i*)(AS3 char*)&ldsT[buf ^ 1][c_cp * 32 + ko_cp];
            __builtin_amdgcn_global_load_async_to_lds_b128(gp, lp, 0, 0);
            __builtin_amdgcn_global_load_async_to_lds_b128(gp, lp, 16, 0);
        }
#else
        uint4 n0 = {0,0,0,0}, n1 = {0,0,0,0};
        if (more) {
            const _Float16* p = gsrc + (size_t)(j + 32);
            n0 = *(const uint4*)(p + 0);
            n1 = *(const uint4*)(p + 8);
        }
        __syncthreads();   // tile `buf` visible; prev readers of buf^1 done
#endif

        __builtin_prefetch(adjrow + j + 1024, 0, 1);   // stream adj ahead

        int4   a0 = *(const int4*)  (adjrow + j + kbA);
        int4   a1 = *(const int4*)  (adjrow + j + kbA + 4);
        int4   a2 = *(const int4*)  (adjrow + j + kbA + 8);
        int4   a3 = *(const int4*)  (adjrow + j + kbA + 12);
        float4 d0 = *(const float4*)(dst    + j + kbA);
        float4 d1 = *(const float4*)(dst    + j + kbA + 4);
        float4 d2 = *(const float4*)(dst    + j + kbA + 8);
        float4 d3 = *(const float4*)(dst    + j + kbA + 12);

        v16h A;
        #define ELEM(e, av, dv) { \
            float x  = leaky(srcv + (dv)) - mval; \
            float pp = __expf(x); \
            pp = ((av) > 0) ? pp : 0.f; \
            sumw += pp; \
            A[e] = (_Float16)pp; }
        ELEM(0,  a0.x, d0.x) ELEM(1,  a0.y, d0.y) ELEM(2,  a0.z, d0.z) ELEM(3,  a0.w, d0.w)
        ELEM(4,  a1.x, d1.x) ELEM(5,  a1.y, d1.y) ELEM(6,  a1.z, d1.z) ELEM(7,  a1.w, d1.w)
        ELEM(8,  a2.x, d2.x) ELEM(9,  a2.y, d2.y) ELEM(10, a2.z, d2.z) ELEM(11, a2.w, d2.w)
        ELEM(12, a3.x, d3.x) ELEM(13, a3.y, d3.y) ELEM(14, a3.z, d3.z) ELEM(15, a3.w, d3.w)
        #undef ELEM

        #pragma unroll
        for (int t = 0; t < 4; ++t) {
            v16h B = *(const v16h*)&ldsT[buf][(t * 16 + m) * 32 + kbB];
            acc[t] = __builtin_amdgcn_wmma_f32_16x16x32_f16(
                false, A, false, B, (short)0, acc[t], false, false);
        }

#if USE_ASYNC_LDS
        __builtin_amdgcn_s_wait_asynccnt(0);   // next tile landed
        __syncthreads();                       // all waves done reading `buf`
#else
        if (more) {
            uint4* d = (uint4*)&ldsT[buf ^ 1][c_cp * 32 + ko_cp];
            d[0] = n0; d[1] = n1;
        }
#endif
        buf ^= 1;
    }

    // Row softmax denominators: lane l and l^16 cover complementary K-halves.
    sumw += __shfl_xor(sumw, 16, 32);
    if (lane < 16) ldsSum[wave][m] = sumw;
    __syncthreads();

    const int mhi = (lane < 16) ? 0 : 8;
    #pragma unroll
    for (int t = 0; t < 4; ++t) {
        #pragma unroll
        for (int r = 0; r < 8; ++r) {
            int M = r + mhi;
            float denom = ldsSum[wave][M];
            float v = acc[t][r] / denom;
            v = (v > 0.f) ? v : (__expf(v) - 1.f);   // elu
            hout[(size_t)layer * GN * GC + (size_t)(rowBase + M) * GC + t * 16 + m] = v;
        }
    }
}

// ---------------------------------------------------------------------------
// Kernel 5: per-block partial NLL over the 1024 training rows (deterministic).
// ---------------------------------------------------------------------------
__global__ void loss_kernel(const float* __restrict__ h, const int* __restrict__ labels,
                            const int* __restrict__ idx, float* __restrict__ partials) {
    __shared__ float red[256];
    int s = blockIdx.x * 256 + threadIdx.x;       // 0..1023
    int i = idx[s];
    const float* r1 = h + (size_t)i * GC;
    const float* r2 = h + (size_t)GN * GC + (size_t)i * GC;

    float mx = -3.402823e38f;
    #pragma unroll
    for (int c = 0; c < GC; ++c) {
        float x = 0.5f * (fmaxf(r1[c], 0.f) + fmaxf(r2[c], 0.f));
        mx = fmaxf(mx, x);
    }
    float sum = 0.f;
    #pragma unroll
    for (int c = 0; c < GC; ++c) {
        float x = 0.5f * (fmaxf(r1[c], 0.f) + fmaxf(r2[c], 0.f));
        sum += __expf(x - mx);
    }
    int y = labels[i];
    float xy = 0.5f * (fmaxf(r1[y], 0.f) + fmaxf(r2[y], 0.f));
    float val = (__logf(sum) + mx) - xy;

    red[threadIdx.x] = val;
    __syncthreads();
    for (int st = 128; st > 0; st >>= 1) {
        if (threadIdx.x < st) red[threadIdx.x] += red[threadIdx.x + st];
        __syncthreads();
    }
    if (threadIdx.x == 0) partials[blockIdx.x] = red[0];
}

__global__ void loss_final(const float* __restrict__ partials, float* __restrict__ out) {
    out[0] = (partials[0] + partials[1] + partials[2] + partials[3]) * (1.0f / 1024.0f);
}

// ---------------------------------------------------------------------------
extern "C" void kernel_launch(void* const* d_in, const int* in_sizes, int n_in,
                              void* d_out, int out_size, void* d_ws, size_t ws_size,
                              hipStream_t stream) {
    const float* F      = (const float*)d_in[0];   // [8192,512]
    const float* W      = (const float*)d_in[1];   // [512,64]
    const float* a      = (const float*)d_in[2];   // [128]
    const int*   adj1   = (const int*)d_in[3];     // [8192,8192]
    const int*   adj2   = (const int*)d_in[4];     // [8192,8192]
    const int*   labels = (const int*)d_in[5];     // [8192]
    const int*   idx    = (const int*)d_in[6];     // [1024]
    float* out = (float*)d_out;

    // Workspace layout (~7.2 MB total)
    float*     Wh       = (float*)d_ws;                       // 524288 f32 (2MB)
    _Float16*  WhT      = (_Float16*)(Wh + GN * GC);          // 524288 f16 (1MB)
    _Float16*  Wt       = WhT + GN * GC;                      // 32768 f16 (64KB)
    float*     src      = (float*)(Wt + GC * GFT);            // 8192 f32
    float*     dst      = src + GN;                           // 8192 f32
    float*     gmax     = dst + GN;                           // 1 f32 (+pad)
    float*     h        = gmax + 16;                          // 2 x 524288 f32 (4MB)
    float*     partials = h + 2 * GN * GC;                    // 4 f32

    wt_kernel<<<GC, 256, 0, stream>>>(W, Wt);
    wh_kernel<<<GN / 16, 32, 0, stream>>>(F, Wt, Wh, WhT);
    srcdst_kernel<<<GN / 256, 256, 0, stream>>>(Wh, a, src, dst);
    gmax_kernel<<<1, 256, 0, stream>>>(dst, gmax);
    dim3 grid(GN / 64, 2);
    attn_kernel<<<grid, 128, 0, stream>>>(adj1, adj2, WhT, src, dst, gmax, h);
    loss_kernel<<<4, 256, 0, stream>>>(h, labels, idx, partials);
    loss_final<<<1, 1, 0, stream>>>(partials, out);
}
